// RecurrentEmbedding_76364518523412
// MI455X (gfx1250) — compile-verified, matrix operand order checked by reading, not run on hardware
//
#include <hip/hip_runtime.h>
#include <hip/hip_bf16.h>

// ---------------- problem constants ----------------
#define B_N   128
#define T_N   512
#define D_N   300          // input features
#define U_N   300          // gru units
#define G_N   900          // 3*U
#define DN_N  100          // dense units
#define BT_N  (B_N * T_N)  // 65536

// padded GEMM dims (K multiple of 32 for 16x16x32 bf16 WMMA, N multiple of 16)
#define KX    320          // padded D and padded U (K dims)
#define NP    960          // padded 3U (N dim of proj / recurrent GEMMs)
#define RECS  912          // rec buffer stride (57 n-tiles * 16)
#define KD    640          // padded 2U (dense K)
#define NDP   112          // padded DENSE (7 n-tiles)
#define NT_G  57           // ceil(900/16)

// scan decomposition: 10 workgroups per direction, each owns 6 n-tiles of Ut
#define NWGD    10
#define NT_LOC  6
#define LROWS   (NT_LOC * 16)   // 96 rows of Ut staged in LDS (60 KB bf16)

typedef __attribute__((ext_vector_type(16))) __bf16 v16bf;
typedef __attribute__((ext_vector_type(8)))  __bf16 bf16x8;
typedef __attribute__((ext_vector_type(8)))  float  v8f;
typedef __attribute__((ext_vector_type(4)))  unsigned u32x4;
typedef __attribute__((ext_vector_type(8)))  int      i32x8;
typedef __attribute__((ext_vector_type(4)))  int      i32x4;

#if __has_builtin(__builtin_amdgcn_tensor_load_to_lds)
#define HAVE_TDM 1
#else
#define HAVE_TDM 0
#endif

// Build the 16x32 bf16 A/B operand for one lane: VGPR0-3 = 8 contiguous K
// elements, VGPR4-7 = the next 8 (K+16 group), per the CDNA5 ISA layout.
// Lowers to two {global,ds}_load_b128.
__device__ inline v16bf ld16(const __bf16* p) {
  bf16x8 lo = *(const bf16x8*)(p);
  bf16x8 hi = *(const bf16x8*)(p + 16);
  v16bf r;
#pragma unroll
  for (int i = 0; i < 8; ++i) { r[i] = lo[i]; r[i + 8] = hi[i]; }
  return r;
}

__device__ inline float sigmoidf_(float v) {
  return 1.0f / (1.0f + __expf(-v));
}
__device__ inline float tanhf_(float v) {
  v = fminf(fmaxf(v, -15.0f), 15.0f);
  float e = __expf(-2.0f * v);
  return (1.0f - e) / (1.0f + e);
}

// ---------------- prep kernels ----------------

// x [BT,300] fp32 -> xb [BT,320] bf16 (zero pad); mask[bt] = any(x != 0)
__global__ void k_prep_x(const float* __restrict__ x, __bf16* __restrict__ xb,
                         unsigned char* __restrict__ mask) {
  int bt = blockIdx.x * blockDim.x + threadIdx.x;
  if (bt >= BT_N) return;
  const float* xr = x + (long)bt * D_N;
  __bf16* xo = xb + (long)bt * KX;
  int any = 0;
  for (int k = 0; k < KX; ++k) {
    float v = (k < D_N) ? xr[k] : 0.0f;
    any |= (v != 0.0f);
    xo[k] = (__bf16)v;
  }
  mask[bt] = (unsigned char)any;
}

// transpose + pad: dst[n*kp + k] = (n<cols && k<rows) ? src[k*cols+n] : 0
__global__ void k_prep_wt(const float* __restrict__ src, __bf16* __restrict__ dst,
                          int rows, int cols, int kp, long total) {
  for (long i = (long)blockIdx.x * blockDim.x + threadIdx.x; i < total;
       i += (long)gridDim.x * blockDim.x) {
    int n = (int)(i / kp);
    int k = (int)(i - (long)n * kp);
    float v = (n < cols && k < rows) ? src[(long)k * cols + n] : 0.0f;
    dst[i] = (__bf16)v;
  }
}

__global__ void k_zero(unsigned char* p, unsigned long n) {
  for (unsigned long i = (unsigned long)blockIdx.x * blockDim.x + threadIdx.x;
       i < n; i += (unsigned long)gridDim.x * blockDim.x)
    p[i] = 0;
}

// zero the pad columns (600..639) of the concat hidden buffer every launch
__global__ void k_ys_pad(__bf16* __restrict__ ysb) {
  long total = (long)BT_N * (KD - 2 * U_N);
  long i = (long)blockIdx.x * blockDim.x + threadIdx.x;
  if (i >= total) return;
  long bt = i / (KD - 2 * U_N);
  int j = (int)(i - bt * (KD - 2 * U_N));
  ysb[bt * KD + 2 * U_N + j] = (__bf16)0.0f;
}

// ---------------- input-projection GEMM: xg = x @ W + b0 ----------------
// one wave computes a 16x64 strip (A reused across 4 n-tiles); grid (512, 15)
__global__ __launch_bounds__(256) void k_xg(const __bf16* __restrict__ xb,
                                            const __bf16* __restrict__ Wt,
                                            const float* __restrict__ b0,
                                            float* __restrict__ xg) {
  int wave = threadIdx.x >> 5, lane = threadIdx.x & 31;
  int lm = lane & 15, kg = lane >> 4;
  int mt = blockIdx.x * 8 + wave;   // 0..4095
  int ng = blockIdx.y;              // 0..14 (group of 4 n-tiles)
  const __bf16* arow = xb + (long)(mt * 16 + lm) * KX + kg * 8;
  const __bf16* brow = Wt + (long)(ng * 64 + lm) * KX + kg * 8;
  v8f acc0 = {}, acc1 = {}, acc2 = {}, acc3 = {};
#pragma unroll
  for (int c = 0; c < 10; ++c) {
    v16bf a = ld16(arow + c * 32);
    v16bf b0v = ld16(brow + c * 32);
    v16bf b1v = ld16(brow + (long)16 * KX + c * 32);
    v16bf b2v = ld16(brow + (long)32 * KX + c * 32);
    v16bf b3v = ld16(brow + (long)48 * KX + c * 32);
    acc0 = __builtin_amdgcn_wmma_f32_16x16x32_bf16(false, a, false, b0v, (short)0, acc0, false, false);
    acc1 = __builtin_amdgcn_wmma_f32_16x16x32_bf16(false, a, false, b1v, (short)0, acc1, false, false);
    acc2 = __builtin_amdgcn_wmma_f32_16x16x32_bf16(false, a, false, b2v, (short)0, acc2, false, false);
    acc3 = __builtin_amdgcn_wmma_f32_16x16x32_bf16(false, a, false, b3v, (short)0, acc3, false, false);
  }
  v8f accs[4] = {acc0, acc1, acc2, acc3};
#pragma unroll
  for (int j = 0; j < 4; ++j) {
    int n = ng * 64 + j * 16 + lm;
    float bias = (n < G_N) ? b0[n] : 0.0f;
    float* orow = xg + (long)(mt * 16 + kg * 8) * NP + n;
#pragma unroll
    for (int v = 0; v < 8; ++v) orow[(long)v * NP] = accs[j][v] + bias;
  }
}

// ---------------- persistent GRU scan (both directions) ----------------

__device__ inline void grid_barrier(unsigned* cnt, unsigned* gen, unsigned nwg) {
  __syncthreads();
  if (threadIdx.x == 0) {
    unsigned g = __hip_atomic_load(gen, __ATOMIC_ACQUIRE, __HIP_MEMORY_SCOPE_AGENT);
    unsigned prev = __hip_atomic_fetch_add(cnt, 1u, __ATOMIC_ACQ_REL,
                                           __HIP_MEMORY_SCOPE_AGENT);
    if (prev == nwg - 1u) {
      __hip_atomic_store(cnt, 0u, __ATOMIC_RELAXED, __HIP_MEMORY_SCOPE_AGENT);
      __hip_atomic_fetch_add(gen, 1u, __ATOMIC_ACQ_REL, __HIP_MEMORY_SCOPE_AGENT);
    } else {
      while (__hip_atomic_load(gen, __ATOMIC_ACQUIRE,
                               __HIP_MEMORY_SCOPE_AGENT) == g)
        __builtin_amdgcn_s_sleep(2);
    }
  }
  __syncthreads();
}

// grid = 2*NWGD workgroups x 512 threads; blocks [0,NWGD) = fwd, rest = bwd.
// Each WG owns n-tiles [wg*6, wg*6+6) and stages those 96 Ut rows in LDS.
__global__ __launch_bounds__(512) void k_scan(
    const float* __restrict__ xg,         // [2][BT][NP]
    const __bf16* __restrict__ Ut,        // [2][NP][KX]
    const float* __restrict__ bfv,        // [2][900] (fwd biases)
    const float* __restrict__ bbv,        // [2][900] (bwd biases)
    const unsigned char* __restrict__ mask,
    float* __restrict__ hbuf,             // [2][B][300] fp32 carry
    __bf16* __restrict__ hb16,            // [2][B][KX] bf16 WMMA operand
    float* __restrict__ rec,              // [2][B][RECS]
    __bf16* __restrict__ ysb,             // [BT][KD] concat bf16
    unsigned* __restrict__ bar) {         // [2][4]
  const int dir = (blockIdx.x >= NWGD) ? 1 : 0;
  const int wg  = blockIdx.x - dir * NWGD;
  const float* xgd = xg + (long)dir * BT_N * NP;
  const __bf16* Utd = Ut + (long)dir * NP * KX;
  const float* b1 = (dir ? bbv : bfv) + G_N;   // recurrent bias row
  float* hb = hbuf + (long)dir * B_N * U_N;
  __bf16* h16 = hb16 + (long)dir * B_N * KX;
  float* recd = rec + (long)dir * B_N * RECS;
  unsigned* bd = bar + dir * 4;

  const int wave = threadIdx.x >> 5, lane = threadIdx.x & 31;
  const int lm = lane & 15, kg = lane >> 4;
  const int tid = wg * 512 + threadIdx.x;       // 0..5119
  const int ntloc = (NT_G - wg * NT_LOC) < NT_LOC ? (NT_G - wg * NT_LOC) : NT_LOC;

  // ---- stage this WG's Ut slice (96 rows x 320 K, bf16 = 60 KB) into LDS ----
  __shared__ __align__(16) __bf16 Bs[LROWS * KX];
#if HAVE_TDM
  if (threadIdx.x == 0) {
    // 2D TDM descriptor: tile 96 rows x 320 elems, elem = 2 bytes, row-major.
    unsigned lds_off =
        (unsigned)(unsigned long long)(__attribute__((address_space(3))) __bf16*)Bs;
    unsigned long long ga =
        (unsigned long long)(const void*)(Utd + (long)wg * LROWS * KX);
    u32x4 g0;
    g0[0] = 1u;                                   // count=1, user descriptor
    g0[1] = lds_off;                              // lds_addr (bytes)
    g0[2] = (unsigned)(ga & 0xffffffffu);         // global_addr lo
    g0[3] = (unsigned)((ga >> 32) & 0x01ffffffu) | 0x80000000u; // hi | type=2
    unsigned dim1 = (unsigned)(NP - wg * LROWS);  // rows remaining from tile
    i32x8 g1;
    g1[0] = (int)(1u << 16);                      // data_size = 2 bytes
    g1[1] = (int)((unsigned)KX << 16);            // tensor_dim0[15:0]
    g1[2] = (int)((dim1 & 0xffffu) << 16);        // tensor_dim1[15:0]
    g1[3] = (int)((unsigned)KX << 16);            // tile_dim0 = 320
    g1[4] = (int)LROWS;                           // tile_dim1 = 96
    g1[5] = (int)KX;                              // tensor_dim0_stride = 320
    g1[6] = 0;
    g1[7] = 0;
    i32x4 z4 = {0, 0, 0, 0};
    i32x8 z8 = {0, 0, 0, 0, 0, 0, 0, 0};
    __builtin_amdgcn_tensor_load_to_lds(g0, g1, z4, z4, z8, 0);
    __builtin_amdgcn_s_wait_tensorcnt(0);
  }
#else
  for (int i = threadIdx.x; i < LROWS * (KX / 8); i += 512) {
    int r = i / (KX / 8), cb = i - r * (KX / 8);
    *(bf16x8*)&Bs[r * KX + cb * 8] =
        *(const bf16x8*)&Utd[((long)wg * LROWS + r) * KX + cb * 8];
  }
#endif
  __syncthreads();

  for (int s = 0; s < T_N; ++s) {
    // stage 1: rec tiles for this WG's n-range (<=48 tiles over 16 waves),
    // B operand read from LDS, A (h state) from global/L2.
    for (int tile = wave; tile < ntloc * 8; tile += 16) {
      int mt = tile & 7, ntl = tile >> 3;
      const __bf16* ar = h16 + (long)(mt * 16 + lm) * KX + kg * 8;
      const __bf16* br = &Bs[(ntl * 16 + lm) * KX + kg * 8];
      v8f acc = {};
#pragma unroll
      for (int c = 0; c < 10; ++c) {
        v16bf a = ld16(ar + c * 32);
        v16bf b = ld16(br + c * 32);
        acc = __builtin_amdgcn_wmma_f32_16x16x32_bf16(false, a, false, b,
                                                      (short)0, acc, false, false);
      }
      int n = (wg * NT_LOC + ntl) * 16 + lm;
      float bias = (n < G_N) ? b1[n] : 0.0f;
      float* orow = recd + (long)(mt * 16 + kg * 8) * RECS + n;
#pragma unroll
      for (int v = 0; v < 8; ++v) orow[(long)v * RECS] = acc[v] + bias;
    }
    grid_barrier(bd, bd + 1, NWGD);

    // stage 2: gates (fp32), masked carry, write bf16 state + concat output
    int t = dir ? (T_N - 1 - s) : s;
    for (int e = tid; e < B_N * U_N; e += NWGD * 512) {
      int b = e / U_N, u = e - b * U_N;
      long bt = (long)b * T_N + t;
      float xz = xgd[bt * NP + u];
      float xr = xgd[bt * NP + u + U_N];
      float xh = xgd[bt * NP + u + 2 * U_N];
      float rz = recd[(long)b * RECS + u];
      float rr = recd[(long)b * RECS + u + U_N];
      float rh = recd[(long)b * RECS + u + 2 * U_N];
      float hp = hb[e];
      float z = sigmoidf_(xz + rz);
      float r = sigmoidf_(xr + rr);
      float hh = tanhf_(xh + r * rh);
      float hn = z * hp + (1.0f - z) * hh;
      if (!mask[bt]) hn = hp;
      hb[e] = hn;
      h16[(long)b * KX + u] = (__bf16)hn;
      ysb[bt * KD + dir * U_N + u] = (__bf16)hn;
    }
    grid_barrier(bd + 2, bd + 3, NWGD);
  }
}

// ---------------- dense: out = tanh(ys @ Wd + bd) ----------------
__global__ __launch_bounds__(256) void k_dense(const __bf16* __restrict__ ysb,
                                               const __bf16* __restrict__ Wdt,
                                               const float* __restrict__ bdv,
                                               float* __restrict__ out) {
  int wave = threadIdx.x >> 5, lane = threadIdx.x & 31;
  int lm = lane & 15, kg = lane >> 4;
  int mt = blockIdx.x * 8 + wave;   // 0..4095
  const __bf16* arow = ysb + (long)(mt * 16 + lm) * KD + kg * 8;
  for (int nt = 0; nt < NDP / 16; ++nt) {
    const __bf16* brow = Wdt + (long)(nt * 16 + lm) * KD + kg * 8;
    v8f acc = {};
#pragma unroll
    for (int c = 0; c < 20; ++c) {
      v16bf a = ld16(arow + c * 32);
      v16bf b = ld16(brow + c * 32);
      acc = __builtin_amdgcn_wmma_f32_16x16x32_bf16(false, a, false, b,
                                                    (short)0, acc, false, false);
    }
    int n = nt * 16 + lm;
    if (n < DN_N) {
      float bias = bdv[n];
      float* orow = out + (long)(mt * 16 + kg * 8) * DN_N + n;
#pragma unroll
      for (int v = 0; v < 8; ++v) orow[(long)v * DN_N] = tanhf_(acc[v] + bias);
    }
  }
}

// ---------------- host side ----------------
extern "C" void kernel_launch(void* const* d_in, const int* in_sizes, int n_in,
                              void* d_out, int out_size, void* d_ws, size_t ws_size,
                              hipStream_t stream) {
  const float* x  = (const float*)d_in[0];
  const float* Wf = (const float*)d_in[1];
  const float* Uf = (const float*)d_in[2];
  const float* bf = (const float*)d_in[3];
  const float* Wb = (const float*)d_in[4];
  const float* Ub = (const float*)d_in[5];
  const float* bb = (const float*)d_in[6];
  const float* Wd = (const float*)d_in[7];
  const float* bd = (const float*)d_in[8];
  float* out = (float*)d_out;

  // workspace layout
  char* base = (char*)d_ws;
  size_t off = 0;
  auto take = [&](size_t bytes) -> char* {
    char* p = base + off;
    off = (off + bytes + 255) & ~(size_t)255;
    return p;
  };
  __bf16* xb   = (__bf16*)take((size_t)BT_N * KX * 2);
  float*  xg   = (float*) take((size_t)2 * BT_N * NP * 4);
  __bf16* Wt   = (__bf16*)take((size_t)2 * NP * KX * 2);   // [fwd][bwd]
  __bf16* Ut   = (__bf16*)take((size_t)2 * NP * KX * 2);
  __bf16* Wdt  = (__bf16*)take((size_t)NDP * KD * 2);
  float*  rec  = (float*) take((size_t)2 * B_N * RECS * 4);
  float*  hbuf = (float*) take((size_t)2 * B_N * U_N * 4);
  __bf16* hb16 = (__bf16*)take((size_t)2 * B_N * KX * 2);
  __bf16* ysb  = (__bf16*)take((size_t)BT_N * KD * 2);
  unsigned char* mask = (unsigned char*)take((size_t)BT_N);
  unsigned* bar = (unsigned*)take(16 * 4);
  (void)ws_size; (void)n_in; (void)in_sizes; (void)out_size;

  // prep: bf16 conversions, transposed/padded weights, mask
  k_prep_x<<<BT_N / 256, 256, 0, stream>>>(x, xb, mask);
  long wtot = (long)NP * KX;
  k_prep_wt<<<512, 256, 0, stream>>>(Wf, Wt,            D_N, G_N, KX, wtot);
  k_prep_wt<<<512, 256, 0, stream>>>(Wb, Wt + wtot,     D_N, G_N, KX, wtot);
  k_prep_wt<<<512, 256, 0, stream>>>(Uf, Ut,            U_N, G_N, KX, wtot);
  k_prep_wt<<<512, 256, 0, stream>>>(Ub, Ut + wtot,     U_N, G_N, KX, wtot);
  k_prep_wt<<<256, 256, 0, stream>>>(Wd, Wdt, 2 * U_N, DN_N, KD, (long)NDP * KD);

  // per-launch state init (h0 = 0, pads = 0, barriers = 0)
  k_zero<<<300, 256, 0, stream>>>((unsigned char*)hbuf, (size_t)2 * B_N * U_N * 4);
  k_zero<<<160, 256, 0, stream>>>((unsigned char*)hb16, (size_t)2 * B_N * KX * 2);
  k_zero<<<1, 64, 0, stream>>>((unsigned char*)bar, 64);
  long padtot = (long)BT_N * (KD - 2 * U_N);
  k_ys_pad<<<(unsigned)((padtot + 255) / 256), 256, 0, stream>>>(ysb);

  // bulk input-projection GEMMs (both directions), 16x64 strip per wave
  dim3 gxg(BT_N / 16 / 8, NP / 64);
  k_xg<<<gxg, 256, 0, stream>>>(xb, Wt,        bf,        xg);
  k_xg<<<gxg, 256, 0, stream>>>(xb, Wt + wtot, bb,        xg + (long)BT_N * NP);

  // persistent bidirectional scan (NWGD WGs per direction, LDS-staged Ut)
  k_scan<<<2 * NWGD, 512, 0, stream>>>(xg, Ut, bf, bb, mask, hbuf, hb16, rec,
                                       ysb, bar);

  // dense + tanh
  k_dense<<<BT_N / 16 / 8, 256, 0, stream>>>(ysb, Wdt, bd, out);
}
